// MultiHeadAttention_2164663517738
// MI455X (gfx1250) — matrix operand order repaired and checked by reading, not compiled, not run on hardware
//
#include <hip/hip_runtime.h>

// ---------------------------------------------------------------------------
// MHA forward for MI455X (gfx1250, wave32, WMMA 16x16x32 bf16 / f32 acc)
// H=16 heads, D_MODEL=1024, D_K=D_V=64, B=2, S=2048
// ---------------------------------------------------------------------------

typedef __attribute__((ext_vector_type(16))) __bf16 v16bf;
typedef __attribute__((ext_vector_type(8)))  __bf16 v8bf;
typedef __attribute__((ext_vector_type(8)))  float  v8f;

union FragU { v16bf v; v8bf h[2]; };

static __device__ __forceinline__ v16bf make_frag(v8bf lo, v8bf hi) {
    FragU f; f.h[0] = lo; f.h[1] = hi; return f.v;
}

#define WMMA_BF16(A, B, C) \
    __builtin_amdgcn_wmma_f32_16x16x32_bf16(false, (A), false, (B), (short)0, (C), false, false)

static constexpr int HNUM = 16;
static constexpr int DMOD = 1024;
static constexpr int DHEAD = 64;
static constexpr int BNUM = 2;
static constexpr int SEQ = 2048;
static constexpr int MROWS = BNUM * SEQ;   // 4096

// ---------------------------------------------------------------------------
// Tiled WMMA GEMM: [4096 x 1024] @ [1024 x 1024]
// XBF:     input X is bf16 (else f32, converted on LDS fill)
// OUTMODE: 0 -> write bf16 into [B,H,S,64] head-major layout (QKV projections)
//          1 -> write f32 row-major [M,N]                     (output proj)
// ---------------------------------------------------------------------------
template <bool XBF, int OUTMODE>
__global__ void gemm64(const void* __restrict__ Xv,
                       const float* __restrict__ Wg,
                       void* __restrict__ outv) {
    const int tid  = threadIdx.x;
    const int lane = tid & 31;
    const int wave = tid >> 5;
    const int col  = lane & 15;
    const int hi16 = lane >> 4;      // which 16-lane half

    const int n0 = blockIdx.x * 64;
    const int m0 = blockIdx.y * 64;

    __shared__ __bf16 At[64][40];    // [M][K], row stride 80B (16B aligned)
    __shared__ __bf16 Bt[64][40];    // [N][K] (transposed weight tile)

    const float*  Xf = (const float*)Xv;
    const __bf16* Xb = (const __bf16*)Xv;

    v8f C[2][2];
    #pragma unroll
    for (int i = 0; i < 2; ++i)
        #pragma unroll
        for (int j = 0; j < 2; ++j) C[i][j] = (v8f){};

    const int rb = (wave >> 1) * 32;   // row quadrant base
    const int cb = (wave & 1) * 32;    // col quadrant base

    for (int k0 = 0; k0 < DMOD; k0 += 32) {
        __syncthreads();
        // stage X tile 64x32 (convert f32->bf16 if needed)
        #pragma unroll 4
        for (int i = tid; i < 64 * 32; i += 128) {
            const int r = i >> 5, c = i & 31;
            float xv = XBF ? (float)Xb[(size_t)(m0 + r) * DMOD + k0 + c]
                           : Xf[(size_t)(m0 + r) * DMOD + k0 + c];
            At[r][c] = (__bf16)xv;
        }
        // stage W tile 32x64, transposed to [N][K]
        #pragma unroll 4
        for (int i = tid; i < 32 * 64; i += 128) {
            const int kk = i >> 6, n = i & 63;
            Bt[n][kk] = (__bf16)Wg[(size_t)(k0 + kk) * DMOD + n0 + n];
        }
        __syncthreads();

        // A fragments (16x32, lane=M, K=16h+e contiguous, matching B)
        v16bf a0 = make_frag(*(const v8bf*)&At[rb + col][16 * hi16],
                             *(const v8bf*)&At[rb + col][16 * hi16 + 8]);
        v16bf a1 = make_frag(*(const v8bf*)&At[rb + 16 + col][16 * hi16],
                             *(const v8bf*)&At[rb + 16 + col][16 * hi16 + 8]);
        // B fragments (32x16, lane=N, K=16h+e contiguous, matching A)
        v16bf b0 = make_frag(*(const v8bf*)&Bt[cb + col][16 * hi16],
                             *(const v8bf*)&Bt[cb + col][16 * hi16 + 8]);
        v16bf b1 = make_frag(*(const v8bf*)&Bt[cb + 16 + col][16 * hi16],
                             *(const v8bf*)&Bt[cb + 16 + col][16 * hi16 + 8]);

        C[0][0] = WMMA_BF16(a0, b0, C[0][0]);
        C[0][1] = WMMA_BF16(a0, b1, C[0][1]);
        C[1][0] = WMMA_BF16(a1, b0, C[1][0]);
        C[1][1] = WMMA_BF16(a1, b1, C[1][1]);
    }

    // store (C/D layout: VGPR r -> row r+8h, lane -> col)
    #pragma unroll
    for (int mi = 0; mi < 2; ++mi)
        #pragma unroll
        for (int ni = 0; ni < 2; ++ni)
            #pragma unroll
            for (int r = 0; r < 8; ++r) {
                const int gm = m0 + rb + 16 * mi + r + 8 * hi16;
                const int gn = n0 + cb + 16 * ni + col;
                const float val = C[mi][ni][r];
                if (OUTMODE == 0) {
                    const int b = gm >> 11, s = gm & (SEQ - 1);
                    const int h = gn >> 6,  d = gn & 63;
                    ((__bf16*)outv)[(((size_t)b * HNUM + h) * SEQ + s) * DHEAD + d] =
                        (__bf16)val;
                } else {
                    ((float*)outv)[(size_t)gm * DMOD + gn] = val;
                }
            }
}

// ---------------------------------------------------------------------------
// Flash-attention: block = 64 queries (4 waves x 16-row q-tile), one (b,h).
// Q tiles + mask tile staged in LDS (Q once; mask per 32-key step, transposed
// so each lane grabs its 8 row-bytes with a single ds_load_b64).
// ---------------------------------------------------------------------------
__global__ void __launch_bounds__(128, 1)
attn64(const __bf16* __restrict__ Qg,
       const __bf16* __restrict__ Kg,
       const __bf16* __restrict__ Vg,
       const unsigned char* __restrict__ maskg,
       __bf16* __restrict__ Og) {
    const int tid  = threadIdx.x;
    const int lane = tid & 31;
    const int wave = tid >> 5;
    const int col  = lane & 15;
    const int hi16 = lane >> 4;

    const int bh = blockIdx.y;
    const int bb = bh >> 4;
    const int hh = bh & 15;
    const __bf16* Qp = Qg + (size_t)bh * SEQ * DHEAD;
    const __bf16* Kp = Kg + (size_t)bh * SEQ * DHEAD;
    const __bf16* Vp = Vg + (size_t)bh * SEQ * DHEAD;

    const int q0 = blockIdx.x * 64;
    const int qt = q0 + wave * 16;

    __shared__ __bf16 Kt[32][72];                 // [key][d]   row stride 144B
    __shared__ __bf16 Vt[64][40];                 // [d][key]   row stride 80B
    __shared__ __bf16 Pt[4][16][40];              // per-wave P tile [qrow][key]
    __shared__ __bf16 Qs[4][16][72];              // per-wave Q tile [qrow][d]
    __shared__ __attribute__((aligned(8))) unsigned char Mt[32][72];  // [key][q_local]

    // stage this wave's Q tile into LDS once (wave-private, DS in-order)
    {
        const int qr = lane >> 1, seg = lane & 1;     // 16 rows x 2 segs of 32 halfs
        const __bf16* src = Qp + (size_t)(qt + qr) * DHEAD + seg * 32;
        #pragma unroll
        for (int j = 0; j < 4; ++j)
            *(v8bf*)&Qs[wave][qr][seg * 32 + 8 * j] = *(const v8bf*)(src + 8 * j);
    }

    v8f Oc[4];
    #pragma unroll
    for (int t = 0; t < 4; ++t) Oc[t] = (v8f){};
    float mr[8], lr[8];
    #pragma unroll
    for (int r = 0; r < 8; ++r) { mr[r] = -1e30f; lr[r] = 0.0f; }

    const int mrow_base = wave * 16 + 8 * hi16;   // this lane's 8 q-rows (local)

    for (int kb = 0; kb < SEQ; kb += 32) {
        __syncthreads();
        // stage K tile 32x64 (natural layout, 16B vector copies)
        {
            const int row = tid >> 2, seg = tid & 3;
            const __bf16* src = Kp + (size_t)(kb + row) * DHEAD + seg * 16;
            *(v8bf*)&Kt[row][seg * 16]     = *(const v8bf*)(src);
            *(v8bf*)&Kt[row][seg * 16 + 8] = *(const v8bf*)(src + 8);
        }
        // stage V tile transposed: Vt[d][key]
        #pragma unroll 4
        for (int i = tid; i < 32 * 64; i += 128) {
            const int sr = i >> 6, d = i & 63;
            Vt[d][sr] = Vp[(size_t)(kb + sr) * DHEAD + d];
        }
        // stage mask tile transposed: Mt[key_local][q_local]
        {
            const int qr = tid >> 1, seg = tid & 1;   // 64 rows x 2 segs of 16 bytes
            const unsigned char* msrc =
                maskg + (size_t)(q0 + qr) * SEQ + kb + seg * 16;
            const unsigned long long u0 = *(const unsigned long long*)(msrc);
            const unsigned long long u1 = *(const unsigned long long*)(msrc + 8);
            #pragma unroll
            for (int j = 0; j < 8; ++j) {
                Mt[seg * 16 + j][qr]     = (unsigned char)(u0 >> (8 * j));
                Mt[seg * 16 + 8 + j][qr] = (unsigned char)(u1 >> (8 * j));
            }
        }
        __syncthreads();

        // scores: S = Q @ K^T   (two 16-key tiles, K-dim 64 = 2 chunks)
        v8f s0 = (v8f){}, s1 = (v8f){};
        {
            v16bf aq0 = make_frag(*(const v8bf*)&Qs[wave][col][16 * hi16],
                                  *(const v8bf*)&Qs[wave][col][16 * hi16 + 8]);
            v16bf b00 = make_frag(*(const v8bf*)&Kt[col][16 * hi16],
                                  *(const v8bf*)&Kt[col][16 * hi16 + 8]);
            v16bf b10 = make_frag(*(const v8bf*)&Kt[16 + col][16 * hi16],
                                  *(const v8bf*)&Kt[16 + col][16 * hi16 + 8]);
            s0 = WMMA_BF16(aq0, b00, s0);
            s1 = WMMA_BF16(aq0, b10, s1);
            v16bf aq1 = make_frag(*(const v8bf*)&Qs[wave][col][32 + 16 * hi16],
                                  *(const v8bf*)&Qs[wave][col][32 + 16 * hi16 + 8]);
            v16bf b01 = make_frag(*(const v8bf*)&Kt[col][32 + 16 * hi16],
                                  *(const v8bf*)&Kt[col][32 + 16 * hi16 + 8]);
            v16bf b11 = make_frag(*(const v8bf*)&Kt[16 + col][32 + 16 * hi16],
                                  *(const v8bf*)&Kt[16 + col][32 + 16 * hi16 + 8]);
            s0 = WMMA_BF16(aq1, b01, s0);
            s1 = WMMA_BF16(aq1, b11, s1);
        }

        // lane's mask bytes: 8 contiguous q-rows per score tile -> one b64 each
        const unsigned long long mb0 =
            *(const unsigned long long*)&Mt[col][mrow_base];
        const unsigned long long mb1 =
            *(const unsigned long long*)&Mt[16 + col][mrow_base];

        // online softmax; rows r+8h live in VGPR r across 16 lanes (cols)
        #pragma unroll
        for (int r = 0; r < 8; ++r) {
            float x0 = s0[r] * 0.125f;   // 1/sqrt(64)
            float x1 = s1[r] * 0.125f;
            x0 = ((mb0 >> (8 * r)) & 0xffull) ? -1e30f : x0;
            x1 = ((mb1 >> (8 * r)) & 0xffull) ? -1e30f : x1;

            float tmax = fmaxf(x0, x1);
            #pragma unroll
            for (int off = 1; off < 16; off <<= 1)
                tmax = fmaxf(tmax, __shfl_xor(tmax, off));
            const float mnew = fmaxf(mr[r], tmax);
            const float scale = __expf(mr[r] - mnew);
            mr[r] = mnew;
            const float p0 = __expf(x0 - mnew);
            const float p1 = __expf(x1 - mnew);
            float rs = p0 + p1;
            #pragma unroll
            for (int off = 1; off < 16; off <<= 1)
                rs += __shfl_xor(rs, off);
            lr[r] = lr[r] * scale + rs;
            // rescale O accumulators for this row immediately (no sc[] array)
            Oc[0][r] *= scale;
            Oc[1][r] *= scale;
            Oc[2][r] *= scale;
            Oc[3][r] *= scale;
            Pt[wave][r + 8 * hi16][col]      = (__bf16)p0;
            Pt[wave][r + 8 * hi16][16 + col] = (__bf16)p1;
        }

        // P as A-fragment (wave-private LDS, DS ops in-order within wave)
        v16bf pf = make_frag(*(const v8bf*)&Pt[wave][col][16 * hi16],
                             *(const v8bf*)&Pt[wave][col][16 * hi16 + 8]);
        // O += P @ V  (4 output tiles of 16 d-columns, K = 32 keys)
        #pragma unroll
        for (int t = 0; t < 4; ++t) {
            v16bf bv = make_frag(*(const v8bf*)&Vt[16 * t + col][16 * hi16],
                                 *(const v8bf*)&Vt[16 * t + col][16 * hi16 + 8]);
            Oc[t] = WMMA_BF16(pf, bv, Oc[t]);
        }
    }

    // normalize and store O as bf16 into [b, s, h*64+d]
    float invr[8];
    #pragma unroll
    for (int r = 0; r < 8; ++r) invr[r] = 1.0f / lr[r];
    #pragma unroll
    for (int t = 0; t < 4; ++t)
        #pragma unroll
        for (int r = 0; r < 8; ++r) {
            const int sg = qt + r + 8 * hi16;
            const int d  = 16 * t + col;
            Og[((size_t)bb * SEQ + sg) * DMOD + hh * DHEAD + d] =
                (__bf16)(Oc[t][r] * invr[r]);
        }
}

// ---------------------------------------------------------------------------
// Inputs (setup_inputs order): query, key, val, W_Q, W_K, W_V, W_O, mask
// Workspace: Q/K/V bf16 head-major (8MB each) + O bf16 row-major (8MB) = 32MB
// ---------------------------------------------------------------------------
extern "C" void kernel_launch(void* const* d_in, const int* in_sizes, int n_in,
                              void* d_out, int out_size, void* d_ws, size_t ws_size,
                              hipStream_t stream) {
    const float* query = (const float*)d_in[0];
    const float* key   = (const float*)d_in[1];
    const float* val   = (const float*)d_in[2];
    const float* W_Q   = (const float*)d_in[3];
    const float* W_K   = (const float*)d_in[4];
    const float* W_V   = (const float*)d_in[5];
    const float* W_O   = (const float*)d_in[6];
    const unsigned char* mask = (const unsigned char*)d_in[7];

    char* ws = (char*)d_ws;
    const size_t qkv_bytes = (size_t)MROWS * DMOD * sizeof(__bf16);  // 8 MB
    __bf16* Qws = (__bf16*)(ws);
    __bf16* Kws = (__bf16*)(ws + qkv_bytes);
    __bf16* Vws = (__bf16*)(ws + 2 * qkv_bytes);
    __bf16* Ows = (__bf16*)(ws + 3 * qkv_bytes);

    const dim3 ggrid(DMOD / 64, MROWS / 64);   // (16, 64)
    const dim3 gblk(128);

    gemm64<false, 0><<<ggrid, gblk, 0, stream>>>((const void*)query, W_Q, (void*)Qws);
    gemm64<false, 0><<<ggrid, gblk, 0, stream>>>((const void*)key,   W_K, (void*)Kws);
    gemm64<false, 0><<<ggrid, gblk, 0, stream>>>((const void*)val,   W_V, (void*)Vws);

    const dim3 agrid(SEQ / 64, BNUM * HNUM);   // (32, 32)
    attn64<<<agrid, gblk, 0, stream>>>(Qws, Kws, Vws, mask, Ows);

    gemm64<true, 1><<<ggrid, gblk, 0, stream>>>((const void*)Ows, W_O, d_out);
}
